// MultiHeadAttention_53996328845390
// MI455X (gfx1250) — matrix operand order repaired and checked by reading, not compiled, not run on hardware
//
#include <hip/hip_runtime.h>
#include <hip/hip_bf16.h>
#include <math.h>

// Problem constants (fixed by the reference)
#define B_  8
#define S_  1024
#define H_  12
#define HD_ 64
#define ED_ 768

#define AS1 __attribute__((address_space(1)))
#define AS3 __attribute__((address_space(3)))

#if defined(__has_builtin)
#if __has_builtin(__builtin_amdgcn_global_load_async_to_lds_b128)
#define HAVE_ASYNC_LDS 1
#endif
#endif

typedef __attribute__((ext_vector_type(16))) _Float16 v16h;
typedef __attribute__((ext_vector_type(8)))  _Float16 h8;
typedef __attribute__((ext_vector_type(8)))  float    v8f;
typedef __attribute__((ext_vector_type(4)))  float    f4;
typedef int i4 __attribute__((vector_size(16)));  // int4 for async-LDS builtin

// D = A(16x32 f16) * B(32x16 f16) + C(16x16 f32)
static __device__ __forceinline__ v8f wmma16x16x32(v16h a, v16h b, v8f c) {
  return __builtin_amdgcn_wmma_f32_16x16x32_f16(false, a, false, b, (short)0, c,
                                                false, false);
}

static __device__ __forceinline__ h8 ldh8(const _Float16* p) { return *(const h8*)p; }

static __device__ __forceinline__ v16h cat8(h8 lo, h8 hi) {
  v16h r;
#pragma unroll
  for (int i = 0; i < 8; ++i) { r[i] = lo[i]; r[i + 8] = hi[i]; }
  return r;
}

// Convert 8 contiguous f32 -> h8 (two float4 loads)
static __device__ __forceinline__ h8 cvt8f(const float* p) {
  f4 a = *(const f4*)p;
  f4 b = *(const f4*)(p + 4);
  h8 r;
#pragma unroll
  for (int i = 0; i < 4; ++i) { r[i] = (_Float16)a[i]; r[i + 4] = (_Float16)b[i]; }
  return r;
}

// Convert 16 contiguous f32 -> v16h
static __device__ __forceinline__ v16h cvt16f(const float* p) {
  return cat8(cvt8f(p), cvt8f(p + 8));
}

static __device__ __forceinline__ unsigned int pack2h(float a, float b) {
  union { _Float16 h[2]; unsigned int u; } t;
  t.h[0] = (_Float16)a; t.h[1] = (_Float16)b;
  return t.u;
}

// 16-byte global -> LDS stage: async DMA when available (ASYNCcnt path),
// otherwise load-to-VGPR + ds_store.
static __device__ __forceinline__ void copy16_g2l(_Float16* ldst,
                                                  const _Float16* gsrc) {
#if defined(HAVE_ASYNC_LDS)
  __builtin_amdgcn_global_load_async_to_lds_b128((AS1 i4*)gsrc,
                                                 (AS3 i4*)ldst, 0, 0);
#else
  *(h8*)ldst = *(const h8*)gsrc;
#endif
}

static __device__ __forceinline__ void wait_async_lds() {
#if defined(HAVE_ASYNC_LDS)
#if __has_builtin(__builtin_amdgcn_s_wait_asynccnt)
  __builtin_amdgcn_s_wait_asynccnt(0);
#else
  asm volatile("s_wait_asynccnt 0x0" ::: "memory");
#endif
#endif
}

// ---------------------------------------------------------------------------
// Kernel 0: convert Wq/Wk/Wv [h][d][e] f32 -> Wt [m][h][e][d] f16 (transposed)
// ---------------------------------------------------------------------------
__global__ __launch_bounds__(256) void mha_wconv(const float* __restrict__ Wq,
                                                 const float* __restrict__ Wk,
                                                 const float* __restrict__ Wv,
                                                 _Float16* __restrict__ Wt) {
  const int per = H_ * HD_ * HD_;  // 49152
  int idx = blockIdx.x * 256 + threadIdx.x;
  if (idx >= 3 * per) return;
  int m = idx / per;
  int r = idx - m * per;
  int h = r / (HD_ * HD_);
  int t = r - h * (HD_ * HD_);
  int e = t / HD_;
  int d = t - e * HD_;
  const float* W = (m == 0) ? Wq : (m == 1) ? Wk : Wv;
  Wt[idx] = (_Float16)W[h * (HD_ * HD_) + d * HD_ + e];
}

// ---------------------------------------------------------------------------
// Kernel 1: per-head QKV projection via WMMA.
//  Q,K computed transposed-orientation (A=W^T, B=x^T) -> row-major [s][d] f16
//  V   computed normal-orientation   (A=x,   B=W)    -> transposed [d][s] f16
// Block = 128 threads (4 waves); block handles (b,h, 64 rows of s).
// ---------------------------------------------------------------------------
__global__ __launch_bounds__(128) void mha_qkv(const float* __restrict__ x,
                                               const float* __restrict__ bq,
                                               const float* __restrict__ bk,
                                               const float* __restrict__ bv,
                                               const _Float16* __restrict__ Wt,
                                               _Float16* __restrict__ Q,
                                               _Float16* __restrict__ K,
                                               _Float16* __restrict__ Vt) {
  const int lane  = threadIdx.x & 31;
  const int wv    = threadIdx.x >> 5;
  const int ln    = lane & 15;
  const int g     = lane >> 4;
  const int chunk = blockIdx.x & 15;        // S/64 = 16 chunks
  const int bh    = blockIdx.x >> 4;
  const int h     = bh % H_;
  const int b     = bh / H_;
  const int stile = chunk * 64 + wv * 16;
  const int srow  = stile + ln;

  const _Float16* WtQ = Wt;
  const _Float16* WtK = Wt + H_ * HD_ * HD_;
  const _Float16* WtV = Wt + 2 * H_ * HD_ * HD_;

  const float* xrow = x + ((size_t)(b * S_ + srow)) * ED_ + h * HD_;

  // B operand (x^T: K-dim=d contiguous per column s) for Q/K transposed tiles
  v16h Bx[2];
#pragma unroll
  for (int kk = 0; kk < 2; ++kk)
    Bx[kk] = cat8(cvt8f(xrow + kk * 32 + g * 8), cvt8f(xrow + kk * 32 + g * 8 + 16));

  // A operand (x rows: M=s, K-dim=d) for V normal tiles
  v16h Ax[2];
#pragma unroll
  for (int kk = 0; kk < 2; ++kk) {
    const float* p = xrow + kk * 32 + g * 8;
    Ax[kk] = cat8(cvt8f(p), cvt8f(p + 16));
  }

  // ---- Q and K: D tile (M=e, N=s) -> contiguous row-major [s][e] stores ----
#pragma unroll
  for (int m = 0; m < 2; ++m) {
    const _Float16* W    = m ? WtK : WtQ;
    const float*    bias = m ? bk : bq;
    _Float16*       dstM = m ? K : Q;
#pragma unroll
    for (int et = 0; et < 4; ++et) {
      v8f acc = {};
#pragma unroll
      for (int kk = 0; kk < 2; ++kk) {
        // A = W^T tile: lane row e = et*16+ln, contiguous d
        const _Float16* wrow =
            W + (size_t)h * (HD_ * HD_) + (et * 16 + ln) * HD_ + kk * 32 + g * 8;
        v16h A = cat8(ldh8(wrow), ldh8(wrow + 16));
        acc = wmma16x16x32(A, Bx[kk], acc);
      }
      const float* bp = bias + h * HD_ + et * 16 + 8 * g;
      h8 hv;
#pragma unroll
      for (int r = 0; r < 8; ++r) hv[r] = (_Float16)(acc[r] + bp[r]);
      _Float16* dst =
          dstM + ((size_t)bh * S_ + srow) * HD_ + et * 16 + 8 * g;
      *(h8*)dst = hv;
    }
  }

  // ---- V: D tile (M=s, N=e) -> contiguous transposed [e][s] stores ----
#pragma unroll
  for (int et = 0; et < 4; ++et) {
    v8f acc = {};
#pragma unroll
    for (int kk = 0; kk < 2; ++kk) {
      // B = W: lane column e = et*16+ln, d in two 8-half runs (g*8, g*8+16)
      const _Float16* wrow =
          WtV + (size_t)h * (HD_ * HD_) + (et * 16 + ln) * HD_ + kk * 32 + g * 8;
      v16h Bw = cat8(ldh8(wrow), ldh8(wrow + 16));
      acc = wmma16x16x32(Ax[kk], Bw, acc);
    }
    const float bvv = bv[h * HD_ + et * 16 + ln];
    h8 hv;
#pragma unroll
    for (int r = 0; r < 8; ++r) hv[r] = (_Float16)(acc[r] + bvv);
    _Float16* dst =
        Vt + ((size_t)bh * HD_ + et * 16 + ln) * S_ + stile + 8 * g;
    *(h8*)dst = hv;
  }
}

// ---------------------------------------------------------------------------
// Kernel 2: flash attention (no scaling, per the reference).
// Block = 4 waves sharing one (b,h,64-query chunk). K/V^T streamed through
// LDS in 64-key chunks (8KB K + 8KB V^T, double-buffered, async DMA when
// available) so the 4 waves read each tile from LDS instead of 4x from L2.
// Wave owns 16 query rows (q = lane&15 column of all transposed tiles).
// Per 32-key step: S^T = K(A) x Q^T(B); online softmax in f32; P^T re-tiled
// into a WMMA B operand with 8 shfl_xor(16) pair swaps; O^T += Vt(A) x P^T(B).
// ---------------------------------------------------------------------------
__global__ __launch_bounds__(128) void mha_attn(const _Float16* __restrict__ Q,
                                                const _Float16* __restrict__ K,
                                                const _Float16* __restrict__ Vt,
                                                float* __restrict__ out) {
  __shared__ __align__(16) _Float16 sK[2][64 * HD_];  // [kv_local][d]
  __shared__ __align__(16) _Float16 sV[2][HD_ * 64];  // [e][kv_local]

  const int lane  = threadIdx.x & 31;
  const int wv    = threadIdx.x >> 5;
  const int ln    = lane & 15;
  const int g     = lane >> 4;
  const int chunk = blockIdx.x & 15;
  const int bh    = blockIdx.x >> 4;
  const int h     = bh % H_;
  const int b     = bh / H_;
  const int q     = chunk * 64 + wv * 16 + ln;

  const _Float16* gK = K + (size_t)bh * S_ * HD_;
  const _Float16* gV = Vt + (size_t)bh * HD_ * S_;

  // Cooperative stage of one 64-key chunk into LDS buffer (c&1).
  // K chunk is 8KB contiguous; V^T chunk is 64 rows x 128B (row stride 2KB).
  auto issue_chunk = [&](int c) {
    const int buf = c & 1;
    const int kv0 = c * 64;
    const int t   = (int)threadIdx.x;
    const _Float16* gkc = gK + (size_t)kv0 * HD_;
#pragma unroll
    for (int i = 0; i < 4; ++i) {
      const int off = (i * 128 + t) * 8;  // halves, 16B units
      copy16_g2l(&sK[buf][off], gkc + off);
    }
#pragma unroll
    for (int i = 0; i < 4; ++i) {
      const int u  = i * 128 + t;
      const int e  = u >> 3;
      const int c8 = (u & 7) * 8;
      copy16_g2l(&sV[buf][e * 64 + c8], gV + (size_t)e * S_ + kv0 + c8);
    }
  };

  // Q^T B-operand, register-resident for the whole pass
  const _Float16* qrow = Q + ((size_t)bh * S_ + q) * HD_;
  v16h Bq[2];
#pragma unroll
  for (int kk = 0; kk < 2; ++kk) {
    const _Float16* p = qrow + kk * 32 + g * 8;
    Bq[kk] = cat8(ldh8(p), ldh8(p + 16));
  }

  issue_chunk(0);

  float m = -INFINITY;
  float lsum = 0.f;
  v8f O[4] = {};

  for (int c = 0; c < S_ / 64; ++c) {
    wait_async_lds();
    __syncthreads();                          // buffer c ready for all waves
    if (c + 1 < S_ / 64) issue_chunk(c + 1);  // overlap DMA with compute
    const int cur = c & 1;

#pragma unroll
    for (int sub = 0; sub < 64; sub += 32) {
      // S^T tiles: t=0 covers local keys sub..+15, t=1 covers sub+16..+31
      v8f s[2];
#pragma unroll
      for (int t = 0; t < 2; ++t) {
        v8f acc = {};
        const _Float16* krow = &sK[cur][(sub + t * 16 + ln) * HD_];
#pragma unroll
        for (int kk = 0; kk < 2; ++kk) {
          const _Float16* p = krow + kk * 32 + g * 8;
          v16h A = cat8(ldh8(p), ldh8(p + 16));
          acc = wmma16x16x32(A, Bq[kk], acc);
        }
        s[t] = acc;
      }

      // Online softmax over these 32 keys (per q = lane&15)
      float vmax = s[0][0];
#pragma unroll
      for (int r = 0; r < 8; ++r) {
        vmax = fmaxf(vmax, s[0][r]);
        vmax = fmaxf(vmax, s[1][r]);
      }
      vmax = fmaxf(vmax, __shfl_xor(vmax, 16, 32));
      const float mnew  = fmaxf(m, vmax);
      const float scale = __expf(m - mnew);
      m = mnew;

      float p0[8], p1[8], rs = 0.f;
#pragma unroll
      for (int r = 0; r < 8; ++r) {
        p0[r] = __expf(s[0][r] - m);
        p1[r] = __expf(s[1][r] - m);
        rs += p0[r] + p1[r];
      }
      rs += __shfl_xor(rs, 16, 32);
      lsum = lsum * scale + rs;
#pragma unroll
      for (int et = 0; et < 4; ++et)
#pragma unroll
        for (int r = 0; r < 8; ++r) O[et][r] *= scale;

      // Re-tile P^T (C/D layout) into a WMMA B operand:
      // lane needs kv pairs (g*16 + 2j, +1); cross-half via shfl_xor(16)
      unsigned int u0[4], u1[4], x0[4], x1[4];
#pragma unroll
      for (int j = 0; j < 4; ++j) {
        u0[j] = pack2h(p0[2 * j], p0[2 * j + 1]);
        u1[j] = pack2h(p1[2 * j], p1[2 * j + 1]);
      }
#pragma unroll
      for (int j = 0; j < 4; ++j) {
        x0[j] = __shfl_xor(u0[j], 16, 32);
        x1[j] = __shfl_xor(u1[j], 16, 32);
      }
      union { v16h h; unsigned int u[8]; } bp;
#pragma unroll
      for (int j = 0; j < 4; ++j) {
        bp.u[j]     = g ? x1[j] : u0[j];
        bp.u[j + 4] = g ? u1[j] : x0[j];
      }

      // O^T += Vt (A: M=e, K=kv) x P^T (B: K=kv, N=q)
#pragma unroll
      for (int et = 0; et < 4; ++et) {
        const _Float16* vrow = &sV[cur][(et * 16 + ln) * 64 + sub + g * 16];
        v16h A = cat8(ldh8(vrow), ldh8(vrow + 8));
        O[et] = wmma16x16x32(A, bp.h, O[et]);
      }
    }

    __syncthreads();  // all waves done reading buffer c before it is reused
  }

  // Epilogue: divide by row sum, write [b][q][h*64+e] (contiguous in e)
  const float inv = 1.0f / lsum;
  float* orow = out + ((size_t)(b * S_ + q)) * ED_ + h * HD_;
#pragma unroll
  for (int et = 0; et < 4; ++et) {
    f4 lo, hi;
#pragma unroll
    for (int r = 0; r < 4; ++r) {
      lo[r] = O[et][r] * inv;
      hi[r] = O[et][r + 4] * inv;
    }
    float* dst = orow + et * 16 + 8 * g;
    *(f4*)dst       = lo;
    *(f4*)(dst + 4) = hi;
  }
}

// ---------------------------------------------------------------------------
extern "C" void kernel_launch(void* const* d_in, const int* in_sizes, int n_in,
                              void* d_out, int out_size, void* d_ws, size_t ws_size,
                              hipStream_t stream) {
  const float* x  = (const float*)d_in[0];
  const float* Wq = (const float*)d_in[1];
  const float* bq = (const float*)d_in[2];
  const float* Wk = (const float*)d_in[3];
  const float* bk = (const float*)d_in[4];
  const float* Wv = (const float*)d_in[5];
  const float* bv = (const float*)d_in[6];
  float* out = (float*)d_out;

  // Workspace layout (f16): Q | K | V^T | Wt  -> ~38.05 MB total
  const size_t elems = (size_t)B_ * H_ * S_ * HD_;  // 6,291,456
  _Float16* Qh  = (_Float16*)d_ws;
  _Float16* Kh  = Qh + elems;
  _Float16* Vth = Kh + elems;
  _Float16* Wt  = Vth + elems;

  (void)in_sizes; (void)n_in; (void)out_size; (void)ws_size;

  mha_wconv<<<(3 * H_ * HD_ * HD_ + 255) / 256, 256, 0, stream>>>(Wq, Wk, Wv, Wt);
  mha_qkv<<<B_ * H_ * (S_ / 64), 128, 0, stream>>>(x, bq, bk, bv, Wt, Qh, Kh, Vth);
  mha_attn<<<B_ * H_ * (S_ / 64), 128, 0, stream>>>(Qh, Kh, Vth, out);
}
